// UnifiedExpertMoE_31172872635040
// MI455X (gfx1250) — matrix operand order repaired and checked by reading, not compiled, not run on hardware
//
#include <hip/hip_runtime.h>
#include <hip/hip_bf16.h>
#include <math.h>

// Problem constants (from the reference)
#define TOK   4096   // N*P tokens
#define DDIM  1024   // d_model
#define HDIM  2048   // d_inner
#define NEXP  16     // experts

// GEMM tiling
#define BM 128
#define BN 128
#define BK 32
#define LDA 40   // halfs per A row in LDS (32 + 8 pad -> conflict-free b128)
#define LDB 40   // halfs per B col in LDS (transposed)

typedef __attribute__((ext_vector_type(4)))  float        f4;
typedef __attribute__((ext_vector_type(2)))  unsigned int u2;
typedef __attribute__((ext_vector_type(4)))  unsigned int u4;
typedef __attribute__((ext_vector_type(8)))  float        v8f;
typedef __attribute__((ext_vector_type(16))) __bf16       v16bf;

union FragU { u4 u[2]; v16bf v; };

// ---------------------------------------------------------------------------
// Gating: logits -> softmax -> top-2 -> append (token, prob/K) per expert
// One wave per token. Lanes: e = lane&15, half-of-D = lane>>4.
// ---------------------------------------------------------------------------
__global__ __launch_bounds__(256)
void moe_gate_kernel(const float* __restrict__ x,
                     const float* __restrict__ gw,
                     const float* __restrict__ gb,
                     const int*   __restrict__ topk_p,
                     int*   __restrict__ counts,
                     int*   __restrict__ tok_list,
                     float* __restrict__ wgt_list)
{
    const int wave = threadIdx.x >> 5;
    const int lane = threadIdx.x & 31;
    const int t = blockIdx.x * (blockDim.x >> 5) + wave;
    if (t >= TOK) return;

    const int e  = lane & 15;
    const int dh = lane >> 4;                 // 0 or 1: which half of D

    const f4* x4 = (const f4*)(x  + (size_t)t * DDIM + dh * (DDIM / 2));
    const f4* g4 = (const f4*)(gw + (size_t)e * DDIM + dh * (DDIM / 2));

    float acc = 0.f;
#pragma unroll 4
    for (int d = 0; d < DDIM / 8; ++d) {      // 128 f4 iters over half of D
        f4 xv = x4[d], gv = g4[d];
        acc += xv[0] * gv[0] + xv[1] * gv[1] + xv[2] * gv[2] + xv[3] * gv[3];
    }
    acc += __shfl_xor(acc, 16, 32);           // combine the two D-halves
    float logit = acc + gb[e];                // lanes 0..15 hold valid logits

    float l[NEXP];
#pragma unroll
    for (int i = 0; i < NEXP; ++i) l[i] = __shfl(logit, i, 32);

    if (lane == 0) {
        float mx = l[0];
#pragma unroll
        for (int i = 1; i < NEXP; ++i) mx = fmaxf(mx, l[i]);
        float p[NEXP];
        float s = 0.f;
#pragma unroll
        for (int i = 0; i < NEXP; ++i) { p[i] = expf(l[i] - mx); s += p[i]; }
        const float inv = 1.f / s;

        int i1 = 0; float v1 = p[0];
#pragma unroll
        for (int i = 1; i < NEXP; ++i) if (p[i] > v1) { v1 = p[i]; i1 = i; }
        int i2 = (i1 == 0) ? 1 : 0; float v2 = p[i2];
#pragma unroll
        for (int i = 0; i < NEXP; ++i)
            if (i != i1 && p[i] > v2) { v2 = p[i]; i2 = i; }

        const float kscale = 1.f / (float)topk_p[0];
        v1 = v1 * inv * kscale;
        v2 = v2 * inv * kscale;

        int p1 = atomicAdd(&counts[i1], 1);
        tok_list[i1 * TOK + p1] = t;
        wgt_list[i1 * TOK + p1] = v1;
        int p2 = atomicAdd(&counts[i2], 1);
        tok_list[i2 * TOK + p2] = t;
        wgt_list[i2 * TOK + p2] = v2;
    }
}

// ---------------------------------------------------------------------------
// Grouped GEMM: for expert e, out[tok] += w * (x[tok] @ W[e] + b[e])
// grid = (TOK/BM, HDIM/BN, NEXP), block = 256 (8 waves, 4x2 wave grid)
// bf16 WMMA 16x16x32, fp32 accumulate, scatter-add epilogue.
// ---------------------------------------------------------------------------
__global__ __launch_bounds__(256, 1)
void moe_gemm_kernel(const float* __restrict__ x,
                     const float* __restrict__ W,
                     const float* __restrict__ bias,
                     const int*   __restrict__ counts,
                     const int*   __restrict__ tok_list,
                     const float* __restrict__ wgt_list,
                     float*       __restrict__ out)
{
    const int e  = blockIdx.z;
    const int mt = blockIdx.x;
    const int nt = blockIdx.y;
    const int cnt = counts[e];
    if (mt * BM >= cnt) return;               // uniform early exit

    __shared__ __bf16 As[2][BM * LDA];
    __shared__ __bf16 Bs[2][BN * LDB];
    __shared__ int    tokS[BM];
    __shared__ float  wgtS[BM];

    const int tid = threadIdx.x;

    if (tid < BM) {
        int idx = mt * BM + tid;
        int tk = 0; float w = 0.f;
        if (idx < cnt) {
            tk = tok_list[e * TOK + idx];
            w  = wgt_list[e * TOK + idx];
        }
        tokS[tid] = tk;
        wgtS[tid] = w;
    }
    __syncthreads();

    // --- staging thread mappings -------------------------------------------
    // A: thread -> (row 0..127, 16-float half of BK)
    const int arow  = tid >> 1;
    const int ahalf = (tid & 1) * 16;
    // B: thread -> 4(k) x 4(n) sub-block: kg in [0,8), cg in [0,32)
    //    lanes 0..31 cover one k-row in consecutive 16B chunks (coalesced)
    const int bkg = (tid >> 5) * 4;           // k base 0,4,..,28
    const int bcg = (tid & 31) * 4;           // col base 0,4,..,124

    const float* wBase = W + (size_t)e * DDIM * HDIM + (size_t)nt * BN;

    f4 aR[4];        // A: 16 floats of one row-half
    f4 bR[4];        // B: bR[i] = 4 cols at k = bkg + i

    auto fetch = [&](int k0) {
        const f4* ap = (const f4*)(x + (size_t)tokS[arow] * DDIM + k0 + ahalf);
#pragma unroll
        for (int i = 0; i < 4; ++i) aR[i] = ap[i];
#pragma unroll
        for (int i = 0; i < 4; ++i)
            bR[i] = *(const f4*)(wBase + (size_t)(k0 + bkg + i) * HDIM + bcg);
    };

    auto commit = [&](int buf) {
        // A: 16 contiguous halfs -> two 16B LDS stores
        alignas(16) __bf16 t[16];
#pragma unroll
        for (int i = 0; i < 4; ++i)
#pragma unroll
            for (int j = 0; j < 4; ++j) t[i * 4 + j] = (__bf16)aR[i][j];
        u4* dst = (u4*)&As[buf][arow * LDA + ahalf];
        dst[0] = ((const u4*)t)[0];
        dst[1] = ((const u4*)t)[1];
        // B transposed: per column, pack 4 consecutive k halfs -> one b64 store
#pragma unroll
        for (int c = 0; c < 4; ++c) {
            alignas(8) __bf16 q[4];
#pragma unroll
            for (int i = 0; i < 4; ++i) q[i] = (__bf16)bR[i][c];
            *(u2*)&Bs[buf][(bcg + c) * LDB + bkg] = *(const u2*)q;
        }
    };

    // wave tiling: 4 waves along M (32 rows each), 2 along N (64 cols each)
    const int wave  = tid >> 5;
    const int lane  = tid & 31;
    const int waveM = wave & 3;
    const int waveN = wave >> 2;
    const int lrow  = lane & 15;
    const int koffA = (lane < 16) ? 0 : 8;    // 16-bit A layout K offset
    const int koffB = (lane < 16) ? 0 : 8;    // 16-bit B layout K offset

    v8f acc[2][4];
    const v8f zero = {0.f, 0.f, 0.f, 0.f, 0.f, 0.f, 0.f, 0.f};
#pragma unroll
    for (int i = 0; i < 2; ++i)
#pragma unroll
        for (int j = 0; j < 4; ++j) acc[i][j] = zero;

    fetch(0);
    commit(0);
    __syncthreads();

    const int KSTEPS = DDIM / BK;             // 32
    for (int ks = 0; ks < KSTEPS; ++ks) {
        const int cur = ks & 1;
        if (ks + 1 < KSTEPS) fetch((ks + 1) * BK);

        v16bf af[2], bf[4];
#pragma unroll
        for (int i = 0; i < 2; ++i) {
            int m = waveM * 32 + i * 16 + lrow;
            FragU u;
            u.u[0] = *(const u4*)&As[cur][m * LDA + koffA];       // K 0..7 (+koff)
            u.u[1] = *(const u4*)&As[cur][m * LDA + 16 + koffA];  // K 16..23 (+koff)
            af[i] = u.v;
        }
#pragma unroll
        for (int j = 0; j < 4; ++j) {
            int c = waveN * 64 + j * 16 + lrow;
            FragU u;
            const u4* p = (const u4*)&Bs[cur][c * LDB + koffB];
            u.u[0] = p[0];
            u.u[1] = p[2];
            bf[j] = u.v;
        }
#pragma unroll
        for (int i = 0; i < 2; ++i)
#pragma unroll
            for (int j = 0; j < 4; ++j)
                acc[i][j] = __builtin_amdgcn_wmma_f32_16x16x32_bf16(
                    false, af[i], false, bf[j], (short)0, acc[i][j],
                    false, false);

        if (ks + 1 < KSTEPS) commit(cur ^ 1);
        __syncthreads();
    }

    // epilogue: out[tok, n] += w_row * (acc + bias[e, n])
    const int lcol  = lane & 15;
    const int rbase = (lane < 16) ? 0 : 8;    // C/D layout: lanes>=16 hold M+8
#pragma unroll
    for (int i = 0; i < 2; ++i) {
#pragma unroll
        for (int j = 0; j < 4; ++j) {
            const int gn = nt * BN + waveN * 64 + j * 16 + lcol;
            const float bv = bias[e * HDIM + gn];
#pragma unroll
            for (int v = 0; v < 8; ++v) {
                const int r = waveM * 32 + i * 16 + rbase + v;
                const float wv = wgtS[r];
                const int   tk = tokS[r];
                atomicAdd(&out[(size_t)tk * HDIM + gn],
                          (acc[i][j][v] + bv) * wv);
            }
        }
    }
}

// ---------------------------------------------------------------------------
extern "C" void kernel_launch(void* const* d_in, const int* in_sizes, int n_in,
                              void* d_out, int out_size, void* d_ws, size_t ws_size,
                              hipStream_t stream)
{
    const float* x    = (const float*)d_in[0];   // [N,P,D]
    const float* W    = (const float*)d_in[1];   // [E,D,H]
    const float* B    = (const float*)d_in[2];   // [E,H]
    const float* gw   = (const float*)d_in[3];   // [E,D]
    const float* gb   = (const float*)d_in[4];   // [E]
    const int*   topk = (const int*)d_in[5];     // scalar on device

    // workspace layout
    int*   counts   = (int*)d_ws;                                  // 16 ints
    int*   tok_list = (int*)((char*)d_ws + 256);                   // E*TOK ints
    float* wgt_list = (float*)((char*)d_ws + 256 + NEXP * TOK * 4);

    float* out = (float*)d_out;

    hipMemsetAsync(out, 0, (size_t)out_size * sizeof(float), stream);
    hipMemsetAsync(counts, 0, 256, stream);

    // gating: one wave per token, 8 waves per block
    moe_gate_kernel<<<TOK / 8, 256, 0, stream>>>(x, gw, gb, topk,
                                                 counts, tok_list, wgt_list);

    // grouped GEMM with scatter-add
    dim3 grid(TOK / BM, HDIM / BN, NEXP);
    moe_gemm_kernel<<<grid, 256, 0, stream>>>(x, W, B, counts,
                                              tok_list, wgt_list, out);
}
